// BasesDecomposition_29368986370096
// MI455X (gfx1250) — matrix operand order repaired and checked by reading, not compiled, not run on hardware
//
#include <hip/hip_runtime.h>
#include <hip/hip_bf16.h>
#include <stdint.h>

// RGCN basis-decomposition layer for MI455X (gfx1250, wave32, WMMA).
//
//   K1: W[r] = sum_b att[r,b] * bases[b]          (17 x 64 x 64 bf16, [r][e][d])
//   K2: H[n, r, :] = x[n,:] @ W_r  (r<16, bf16) via v_wmma_f32_16x16x32_bf16,
//       D tiles transposed through wave-private LDS -> 128B-coalesced b128 stores.
//       out[n,:]   = mask[n] ? x[n,:] @ W_16 : 0  (fp32, coalesced float4 stores)
//   K3: bidirectional edge gather from H + fp32 global atomic scatter-add.
//
// ws layout: [0 .. 139264)      : Wbf (17*64*64 bf16)
//            [1MB .. 1MB+410MB) : H   (N*16*64 bf16)

#define NN 200000
#define NE 800000
#define DD 64
#define NB 8
#define LPITCH 68   // LDS row pitch in floats: +4 banks/row skew -> conflict-free b128 reads

typedef __attribute__((ext_vector_type(16))) __bf16 v16bf;
typedef __attribute__((ext_vector_type(8)))  __bf16 v8bf;
typedef __attribute__((ext_vector_type(4)))  __bf16 v4bf;
typedef __attribute__((ext_vector_type(8)))  float  v8f;

// ---------------------------------------------------------------- K1: weights
__global__ __launch_bounds__(256) void rgcn_weights(const float* __restrict__ bases,
                                                    const float* __restrict__ att,
                                                    __bf16* __restrict__ Wbf) {
    int tid = blockIdx.x * 256 + threadIdx.x;      // tid = r*4096 + e*64 + d
    if (tid >= 17 * 64 * 64) return;
    int r = tid >> 12;
    int e = (tid >> 6) & 63;
    int d = tid & 63;
    float acc = 0.f;
#pragma unroll
    for (int b = 0; b < NB; ++b)
        acc += att[r * NB + b] * bases[((b << 6) + d) * 64 + e];
    Wbf[tid] = (__bf16)acc;                        // layout [r][e][d], d fastest
}

// helper: 8 consecutive fp32 -> bf16 fragment elements [off..off+7]
__device__ __forceinline__ void load8(const float* __restrict__ p, v16bf& f, int off) {
    float4 u = *(const float4*)p;
    float4 v = *(const float4*)(p + 4);
    f[off + 0] = (__bf16)u.x; f[off + 1] = (__bf16)u.y;
    f[off + 2] = (__bf16)u.z; f[off + 3] = (__bf16)u.w;
    f[off + 4] = (__bf16)v.x; f[off + 5] = (__bf16)v.y;
    f[off + 6] = (__bf16)v.z; f[off + 7] = (__bf16)v.w;
}

// ---------------------------------------------------------------- K2: GEMM
// One wave = one 16-node tile. 8 waves / block. A frags (K=0..31, 32..63) in regs.
__global__ __launch_bounds__(256) void rgcn_gemm(const float* __restrict__ x,
                                                 const uint8_t* __restrict__ keep,
                                                 const __bf16* __restrict__ Wbf,
                                                 __bf16* __restrict__ H,
                                                 float* __restrict__ out) {
    __shared__ __align__(16) float lds[8 * 16 * LPITCH];   // 34 KB, wave-private slices

    int lane = threadIdx.x & 31;
    int wave = threadIdx.x >> 5;
    int tile = blockIdx.x * 8 + wave;
    int base = tile * 16;
    if (base >= NN) return;                         // uniform per wave: EXEC stays full

    float* Lw  = lds + wave * 16 * LPITCH;
    int   lrow = lane & 15;
    int   hi   = lane >> 4;                         // 0 / 1
    const float* xp = x + (size_t)(base + lrow) * DD;

    // A fragment (16x32 bf16), contiguous K: lanes<16 hold K 0..15 (+k0),
    // lanes>=16 hold K 16..31 (+k0)
    v16bf a0, a1;
    int kh = hi * 16;
    load8(xp + kh,           a0, 0);
    load8(xp + kh + 8,       a0, 8);
    load8(xp + 32 + kh,      a1, 0);
    load8(xp + 32 + kh + 8,  a1, 8);

    for (int r = 0; r < 17; ++r) {
        // ---- compute the 16x64 row block for this relation (4 WMMA tiles)
        v8f acc[4];
#pragma unroll
        for (int nt = 0; nt < 4; ++nt) {
            int e = nt * 16 + lrow;                 // output column (one per lane)
            // B fragment (32x16 bf16): lanes<16 -> K 0..15, lanes>=16 -> K 16..31
            const __bf16* wp = Wbf + ((size_t)(r * 64 + e)) * 64 + hi * 16;
            v16bf b0 = *(const v16bf*)(wp);         // K-chunk 0..31
            v16bf b1 = *(const v16bf*)(wp + 32);    // K-chunk 32..63
            v8f c = {};
            c = __builtin_amdgcn_wmma_f32_16x16x32_bf16(
                false, a0, false, b0, (short)0, c, false, false);
            c = __builtin_amdgcn_wmma_f32_16x16x32_bf16(
                false, a1, false, b1, (short)0, c, false, false);
            acc[nt] = c;
        }

        // ---- scatter D layout into wave-private LDS (conflict-free b32 writes)
        // D layout: VGPR j -> row j (lanes<16) / j+8 (lanes>=16), col = nt*16+lrow
#pragma unroll
        for (int nt = 0; nt < 4; ++nt)
#pragma unroll
            for (int j = 0; j < 8; ++j)
                Lw[(hi * 8 + j) * LPITCH + nt * 16 + lrow] = acc[nt][j];

        // wave-private region: only need same-wave DS ordering (DScnt), no barrier
        __asm volatile("s_wait_dscnt 0x0" ::: "memory");

        // ---- read back row-major: lane covers half a node row (32 floats)
        const float* rp = Lw + lrow * LPITCH + hi * 32;
        if (r < 16) {
            __bf16* hp = H + (((size_t)(base + lrow)) * 16 + r) * 64 + hi * 32;
#pragma unroll
            for (int q = 0; q < 4; ++q) {
                float4 f0 = *(const float4*)(rp + q * 8);
                float4 f1 = *(const float4*)(rp + q * 8 + 4);
                v8bf o;
                o[0] = (__bf16)f0.x; o[1] = (__bf16)f0.y;
                o[2] = (__bf16)f0.z; o[3] = (__bf16)f0.w;
                o[4] = (__bf16)f1.x; o[5] = (__bf16)f1.y;
                o[6] = (__bf16)f1.z; o[7] = (__bf16)f1.w;
                *(v8bf*)(hp + q * 8) = o;           // 16B store; lane pair -> 128B row
            }
        } else {
            // self-loop relation -> fp32 out, masked (initializes all of d_out)
            float k = keep[base + lrow] ? 1.0f : 0.0f;
            float* op = out + (size_t)(base + lrow) * DD + hi * 32;
#pragma unroll
            for (int q = 0; q < 8; ++q) {
                float4 f = *(const float4*)(rp + q * 4);
                f.x *= k; f.y *= k; f.z *= k; f.w *= k;
                *(float4*)(op + q * 4) = f;
            }
        }
    }
}

// ---------------------------------------------------------------- K3: scatter
// 16 threads per (virtual) edge, 4 floats each. 2E virtual edges (both directions).
__global__ __launch_bounds__(256) void rgcn_scatter(const __bf16* __restrict__ H,
                                                    const int* __restrict__ src,
                                                    const int* __restrict__ tgt,
                                                    const int* __restrict__ et,
                                                    float* __restrict__ out) {
    long long tid = (long long)blockIdx.x * 256 + threadIdx.x;
    int i = (int)(tid >> 4);
    if (i >= 2 * NE) return;
    int c  = ((int)tid & 15) * 4;
    int ei = (i < NE) ? i : i - NE;
    int s, t;
    if (i < NE) { s = src[ei]; t = tgt[ei]; }
    else        { s = tgt[ei]; t = src[ei]; }
    int r = et[ei];

    v4bf hv = *(const v4bf*)(H + ((size_t)s * 16 + r) * 64 + c);   // 8B, edge = 128B
    float* op = out + (size_t)t * 64 + c;
    unsafeAtomicAdd(op + 0, (float)hv[0]);
    unsafeAtomicAdd(op + 1, (float)hv[1]);
    unsafeAtomicAdd(op + 2, (float)hv[2]);
    unsafeAtomicAdd(op + 3, (float)hv[3]);
}

// ---------------------------------------------------------------- launch
extern "C" void kernel_launch(void* const* d_in, const int* in_sizes, int n_in,
                              void* d_out, int out_size, void* d_ws, size_t ws_size,
                              hipStream_t stream) {
    const float*   x     = (const float*)d_in[0];
    const uint8_t* keep  = (const uint8_t*)d_in[1];
    const int*     src   = (const int*)d_in[2];
    const int*     tgt   = (const int*)d_in[3];
    const int*     et    = (const int*)d_in[4];
    const float*   bases = (const float*)d_in[5];
    const float*   att   = (const float*)d_in[6];
    float*         out   = (float*)d_out;

    __bf16* Wbf = (__bf16*)d_ws;
    __bf16* H   = (__bf16*)((char*)d_ws + (1 << 20));   // needs ~411 MB workspace

    rgcn_weights<<<(17 * 64 * 64 + 255) / 256, 256, 0, stream>>>(bases, att, Wbf);

    int tiles = NN / 16;                                 // 12500 (exact)
    rgcn_gemm<<<(tiles + 7) / 8, 256, 0, stream>>>(x, keep, Wbf, H, out);

    long long sthreads = (long long)2 * NE * 16;         // 25.6M
    rgcn_scatter<<<(int)((sthreads + 255) / 256), 256, 0, stream>>>(H, src, tgt, et, out);
}
